// TorchSimulatorTile_69432441307318
// MI455X (gfx1250) — matrix operand order repaired and checked by reading, not compiled, not run on hardware
//
#include <hip/hip_runtime.h>
#include <cmath>
#include <cstdint>

// CDNA5 WMMA vector types
typedef __attribute__((ext_vector_type(16))) _Float16 v16h;
typedef __attribute__((ext_vector_type(8)))  float    v8f;

// ---- aihwkit IO preset constants ----
#define DAC_LV     127.0f              // round(x*127)/127  == quantize(step 1/127) + clip [-1,1]
#define ADC_STEP   (24.0f / 510.0f)    // OUT_RES * 2 * OUT_BOUND
#define ADC_INV    (510.0f / 24.0f)
#define OUTB       12.0f
#define ONOISE     0.06f

// Block tiling: 128(M) x 256(N) per workgroup, 8 waves of 64x64, BK=32
#define BM 128
#define BN 256
#define BK 32
#define LDH 40                         // 32 + 8 halves pad -> 80B row stride, conflict-free b128

// ------------------------------------------------------------------
__global__ void init_flags_k(int* flags) { flags[0] = 0; flags[1] = 0; }

__global__ __launch_bounds__(256) void row_absmax_k(const float* __restrict__ x,
                                                    float* __restrict__ nm, int K) {
  const int row = blockIdx.x;
  const float* p = x + (size_t)row * K;
  float m = 0.0f;
  for (int k = threadIdx.x; k < K; k += 256) m = fmaxf(m, fabsf(p[k]));
  for (int o = 16; o > 0; o >>= 1) m = fmaxf(m, __shfl_xor(m, o, 32));
  __shared__ float s[8];
  const int lane = threadIdx.x & 31, w = threadIdx.x >> 5;
  if (lane == 0) s[w] = m;
  __syncthreads();
  if (threadIdx.x == 0) {
    float r = s[0];
    #pragma unroll
    for (int i = 1; i < 8; i++) r = fmaxf(r, s[i]);
    nm[row] = (r <= 0.0f) ? 1.0f : r;   // reference: where(nm<=0, 1, nm)
  }
}

// One-time weight conversion f32 -> f16 (W16 stays resident in the 192MB L2)
__global__ __launch_bounds__(256) void convert_w_k(const float* __restrict__ W,
                                                   _Float16* __restrict__ W16) {
  const unsigned idx = (blockIdx.x * 256u + threadIdx.x) * 8u;
  const float4 a = *(const float4*)(W + idx);
  const float4 b = *(const float4*)(W + idx + 4);
  union { _Float16 h[8]; uint4 q; } st;
  st.h[0] = (_Float16)a.x; st.h[1] = (_Float16)a.y;
  st.h[2] = (_Float16)a.z; st.h[3] = (_Float16)a.w;
  st.h[4] = (_Float16)b.x; st.h[5] = (_Float16)b.y;
  st.h[6] = (_Float16)b.z; st.h[7] = (_Float16)b.w;
  *(uint4*)(W16 + idx) = st.q;
}

// Per-pass DAC quantization of the noise/bound-managed input -> f16
__global__ __launch_bounds__(256) void quantize_x_k(const float* __restrict__ X,
                                                    const float* __restrict__ nm,
                                                    _Float16* __restrict__ Xq,
                                                    int K, float red,
                                                    const int* __restrict__ flags) {
  if (flags[0]) return;
  const unsigned idx = (blockIdx.x * 256u + threadIdx.x) * 8u;
  const unsigned m = idx / (unsigned)K;
  const float s = DAC_LV / (nm[m] * red);
  const float4 a = *(const float4*)(X + idx);
  const float4 b = *(const float4*)(X + idx + 4);
  union { _Float16 h[8]; uint4 q; } st;
  const float* f = (const float*)&a;
  #pragma unroll
  for (int j = 0; j < 8; j++) {
    const float v = (j < 4) ? f[j] : ((const float*)&b)[j - 4];
    float q = rintf(v * s);                       // RNE like jnp.round
    q = fminf(fmaxf(q, -DAC_LV), DAC_LV);         // == clip to [-1,1]
    st.h[j] = (_Float16)(q * (1.0f / DAC_LV));
  }
  *(uint4*)(Xq + idx) = st.q;
}

// counter-based RNG (PCG output fn) + Box-Muller
__device__ __forceinline__ unsigned pcg_hash(unsigned x) {
  x = x * 747796405u + 2891336453u;
  x = ((x >> ((x >> 28u) + 4u)) ^ x) * 277803737u;
  return (x >> 22u) ^ x;
}
__device__ __forceinline__ float gauss_at(unsigned idx, unsigned rnd) {
  const unsigned s  = rnd * 0x9E3779B9u;
  const unsigned h1 = pcg_hash(idx * 2u + s);
  const unsigned h2 = pcg_hash(idx * 2u + 1u + s);
  const float u1 = ((float)h1 + 1.0f) * 2.3283064365386963e-10f;  // (0,1]
  const float u2 = (float)h2 * 2.3283064365386963e-10f;
  return sqrtf(-2.0f * __logf(u1)) * __cosf(6.28318530717958647f * u2);
}

// ---- CDNA5 async copy: global -> LDS, 16B per lane, tracked by ASYNCcnt ----
__device__ __forceinline__ void async_b128(const _Float16* g, _Float16* l) {
  const unsigned lo = (unsigned)(unsigned long long)(void*)l;   // LDS offset = low 32 bits
  asm volatile("global_load_async_to_lds_b128 %0, %1, off"
               :: "v"(lo), "v"((const void*)g) : "memory");
}

// stage one 128x32 A tile + 256x32 B tile: 6 async b128 per thread
__device__ __forceinline__ void stage_tile(const _Float16* __restrict__ Xg,
                                           const _Float16* __restrict__ Wg,
                                           _Float16* Ab, _Float16* Bb,
                                           int tid, int kt, int K) {
  // A: 2 threads per row, 2 chunks each
  {
    const int row = tid >> 1;
    const int cb  = (tid & 1) * 16;
    const _Float16* ga = Xg + (size_t)row * K + kt + cb;
    _Float16* la = Ab + row * LDH + cb;
    async_b128(ga,     la);
    async_b128(ga + 8, la + 8);
  }
  // B: 1 thread per row (256 rows), 4 chunks = full 64B row
  {
    const _Float16* gb = Wg + (size_t)tid * K + kt;
    _Float16* lb = Bb + tid * LDH;
    async_b128(gb,      lb);
    async_b128(gb + 8,  lb + 8);
    async_b128(gb + 16, lb + 16);
    async_b128(gb + 24, lb + 24);
  }
}

// ------------------------------------------------------------------
// One analog pass at bound-management factor `red` (pure-f16 WMMA GEMM).
__global__ __launch_bounds__(256) void analog_mvm_pass_k(
    const _Float16* __restrict__ Xq, const _Float16* __restrict__ W16,
    const float* __restrict__ nm, float* __restrict__ out,
    int* __restrict__ flags, int M, int N, int K, float red, int rnd)
{
  if (flags[0]) return;                 // bound management already satisfied (uniform)

  const int tid  = threadIdx.x;
  const int lane = tid & 31;
  const int wave = tid >> 5;
  const int wm   = wave >> 2;           // 0..1  : wave row (64 M each)
  const int wn   = wave & 3;            // 0..3  : wave col (64 N each)
  const int lh   = lane & 15;           // lane-in-half
  const int lg   = lane >> 4;           // half-group (0/1)

  const int m0 = blockIdx.y * BM;
  const int n0 = blockIdx.x * BN;

  __shared__ alignas(16) _Float16 Al[2][BM * LDH];   // 2 x 10KB (padded)
  __shared__ alignas(16) _Float16 Bl[2][BN * LDH];   // 2 x 20KB (padded)

  const _Float16* Xg = Xq  + (size_t)m0 * K;
  const _Float16* Wg = W16 + (size_t)n0 * K;

  v8f acc[4][4];
  #pragma unroll
  for (int mi = 0; mi < 4; mi++)
    #pragma unroll
    for (int ni = 0; ni < 4; ni++) acc[mi][ni] = (v8f)0.0f;

  union Frag { v16h h; uint4 q[2]; };

  stage_tile(Xg, Wg, Al[0], Bl[0], tid, 0, K);       // prologue: tile 0 in flight

  const int NKT = K / BK;
  for (int it = 0; it < NKT; ++it) {
    const int buf = it & 1;
    if (it + 1 < NKT) {
      // issue next tile into the other buffer, then wait for the current one:
      // async loads complete in order, so "<= 6 outstanding" == tile `it` landed
      stage_tile(Xg, Wg, Al[buf ^ 1], Bl[buf ^ 1], tid, (it + 1) * BK, K);
      asm volatile("s_wait_asynccnt 0x6" ::: "memory");
    } else {
      asm volatile("s_wait_asynccnt 0x0" ::: "memory");
    }
    __syncthreads();                                  // whole tile visible to all waves

    // ---- fragments in ISA-documented 16-bit WMMA layouts ----
    const _Float16* Ac = Al[buf];
    const _Float16* Bc = Bl[buf];
    Frag af[4], bf[4];
    #pragma unroll
    for (int mi = 0; mi < 4; mi++) {                  // A: row=lh, K 0-7/16-23 | 8-15/24-31
      const int r = wm * 64 + mi * 16 + lh;
      af[mi].q[0] = *(const uint4*)&Ac[r * LDH + lg * 8];
      af[mi].q[1] = *(const uint4*)&Ac[r * LDH + 16 + lg * 8];
    }
    #pragma unroll
    for (int ni = 0; ni < 4; ni++) {                  // B: col=lh, K 0-7/16-23 | 8-15/24-31
      const int c = wn * 64 + ni * 16 + lh;
      bf[ni].q[0] = *(const uint4*)&Bc[c * LDH + lg * 8];
      bf[ni].q[1] = *(const uint4*)&Bc[c * LDH + 16 + lg * 8];
    }

    #pragma unroll
    for (int mi = 0; mi < 4; mi++)
      #pragma unroll
      for (int ni = 0; ni < 4; ni++)
        acc[mi][ni] = __builtin_amdgcn_wmma_f32_16x16x32_f16(
            false, af[mi].h, false, bf[ni].h, (short)0, acc[mi][ni], false, false);

    __syncthreads();   // all waves done reading `buf` before it is refilled at it+2
  }

  // ---- epilogue: noise -> ADC quantize -> clip detect -> rescale by nm*red ----
  int myclip = 0;
  #pragma unroll
  for (int mi = 0; mi < 4; mi++) {
    #pragma unroll
    for (int ni = 0; ni < 4; ni++) {
      #pragma unroll
      for (int r = 0; r < 8; r++) {
        const int gm = m0 + wm * 64 + mi * 16 + r + lg * 8;   // C layout: M = r + 8*halfgroup
        const int gn = n0 + wn * 64 + ni * 16 + lh;           //           N = lane&15
        float v = acc[mi][ni][r];
        v += ONOISE * gauss_at((unsigned)(gm * (unsigned)N + gn), (unsigned)rnd);
        v = rintf(v * ADC_INV) * ADC_STEP;                    // ADC discretization
        if (v > OUTB || v < -OUTB) myclip = 1;                // checked before saturation
        v = fminf(fmaxf(v, -OUTB), OUTB);                     // ADC saturation
        out[(size_t)gm * N + gn] = v * nm[gm] * red;          // undo input scaling
      }
    }
  }
  if (myclip) atomicOr(flags + 1, 1);
}

// bound-management state machine (graph-safe, no host sync)
__global__ void resolve_k(int* flags, float red) {
  if (!flags[0]) {
    const int clipped = flags[1];
    const int passed = (!clipped) || (red > 1000.0f) || (red > 31.75f);
    flags[0] = passed;
  }
  flags[1] = 0;  // reset clip detector for the next pass
}

// ------------------------------------------------------------------
extern "C" void kernel_launch(void* const* d_in, const int* in_sizes, int n_in,
                              void* d_out, int out_size, void* d_ws, size_t ws_size,
                              hipStream_t stream) {
  const float* X = (const float*)d_in[0];   // [M, K] f32
  const float* W = (const float*)d_in[1];   // [N, K] f32
  float* out = (float*)d_out;               // [M, N] f32

  // in0 = M*K, in1 = N*K, out = M*N  ->  K = sqrt(in0*in1/out)
  const double in0 = (double)in_sizes[0], in1 = (double)in_sizes[1];
  const int K = (int)(sqrt(in0 * in1 / (double)out_size) + 0.5);
  const int M = (int)(in0 / K + 0.5);
  const int N = (int)(in1 / K + 0.5);

  // workspace layout: flags | nm[M] | W16[N*K] | Xq[M*K]
  char* ws = (char*)d_ws;
  int*   flags = (int*)ws;
  size_t off = 64;
  float* nm = (float*)(ws + off);
  off += (size_t)M * 4;       off = (off + 255) & ~(size_t)255;
  _Float16* W16 = (_Float16*)(ws + off);
  off += (size_t)N * K * 2;   off = (off + 255) & ~(size_t)255;
  _Float16* Xq = (_Float16*)(ws + off);

  init_flags_k<<<1, 1, 0, stream>>>(flags);
  row_absmax_k<<<M, 256, 0, stream>>>(X, nm, K);
  convert_w_k<<<(unsigned)((size_t)N * K / 2048), 256, 0, stream>>>(W, W16);

  dim3 grid(N / BN, M / BM);
  const unsigned xblocks = (unsigned)((size_t)M * K / 2048);
  float red = 1.0f;                          // reference: red starts 0.5, doubles in body
  for (int p = 1; p <= 6; p++) {             // red=32 > 31.75 forces termination
    quantize_x_k<<<xblocks, 256, 0, stream>>>(X, nm, Xq, K, red, flags);
    analog_mvm_pass_k<<<grid, 256, 0, stream>>>(Xq, W16, nm, out, flags, M, N, K, red, p);
    resolve_k<<<1, 1, 0, stream>>>(flags, red);
    red *= 2.0f;
  }
}
